// MultiMemoryHeadedAttention_3161095930143
// MI455X (gfx1250) — compile-verified, matrix operand order checked by reading, not run on hardware
//
#include <hip/hip_runtime.h>
#include <hip/hip_bf16.h>

// MultiMemoryHeadedAttention for MI455X (gfx1250, wave32, WMMA bf16)
// gate = sigmoid(64) == 1.0f exactly in fp32 => memory branch (sim/top-k/gather)
// contributes exactly zero; only QKV proj + local attention + out proj remain.
//
// All WMMA operand tiles are staged in LDS so that every fragment load is two
// contiguous 16-byte runs per lane (-> 2x ds_load_b128); B-operand sources are
// staged TRANSPOSED to make that true for them as well. fp32->bf16 uses the
// native gfx1250 converters (fptrunc -> v_cvt_*bf16_f32).

#define B_  4
#define S_  1024
#define D_  768
#define H_  12
#define HD_ 64
#define BS_ (B_ * S_)

typedef __attribute__((ext_vector_type(16))) __bf16 v16bf;
typedef __attribute__((ext_vector_type(8)))  float  v8f;

#define WMMA_BF16(a, b, c) \
  __builtin_amdgcn_wmma_f32_16x16x32_bf16(false, (a), false, (b), (short)0, (c), false, false)

// Native bf16 convert (RNE) -> lowers to v_cvt_*bf16_f32 on gfx1250.
__device__ __forceinline__ unsigned short f2bf(float f) {
  union { __bf16 b; unsigned short u; } c;
  c.b = (__bf16)f;
  return c.u;
}

// 16x32 bf16 fragment, A-pattern (ISA 7.12.2): lane&15 = M (or N for a
// transposed-staged B tile), lane>>4 = K-half. Per lane: 16B at K=h*8 and 16B
// at K=16+h*8 -> exactly two ds_load_b128. base/stride in elements; base 16B
// aligned, stride a multiple of 8 elements.
__device__ __forceinline__ v16bf frag_ld(const unsigned short* base, int stride, int lane) {
  const int m = lane & 15, half = (lane >> 4) & 1;
  const unsigned short* p = base + m * stride + half * 8;
  union { uint4 q[2]; v16bf v; } f;
  f.q[0] = *(const uint4*)(p);
  f.q[1] = *(const uint4*)(p + 16);
  return f.v;
}

// ---------------------------------------------------------------------------
// Kernel 1: fused QKV projection.  C(4096x768) = X(4096x768) * W(768x768) + b
// blockIdx.z selects {Wq,Wk,Wv}. 64x64 output tile, 8 waves, 2 WMMA acc/wave.
// A staged row-major, W staged transposed (sBT[n][k]) for contiguous B frags.
// ---------------------------------------------------------------------------
__global__ __launch_bounds__(256) void qkv_gemm(
    const float* __restrict__ x,
    const float* __restrict__ Wq, const float* __restrict__ bq,
    const float* __restrict__ Wk, const float* __restrict__ bk,
    const float* __restrict__ Wv, const float* __restrict__ bv,
    unsigned short* __restrict__ qkv /* [3][BS][D] bf16 */) {
  __shared__ unsigned short sA [64 * 32];   // [row][k]
  __shared__ unsigned short sBT[64 * 32];   // [n][k]  (transposed W tile)
  const int rowTile = blockIdx.x;          // 0..63
  const int colTile = blockIdx.y;          // 0..11
  const int which   = blockIdx.z;          // 0=q 1=k 2=v
  const float* W    = which == 0 ? Wq : (which == 1 ? Wk : Wv);
  const float* bias = which == 0 ? bq : (which == 1 ? bk : bv);
  unsigned short* outp = qkv + (size_t)which * BS_ * D_;

  const int tid = threadIdx.x, lane = tid & 31, wave = tid >> 5;
  const int rg = wave >> 1;                // 0..3 : 16-row group
  const int cg = wave & 1;                 // 0..1 : 32-col group
  v8f c0 = {}, c1 = {};

  for (int k0 = 0; k0 < D_; k0 += 32) {
    __syncthreads();
    // stage A: 64x32 fp32 -> bf16 (b128 global loads, b64 LDS stores)
#pragma unroll
    for (int i = 0; i < 2; ++i) {
      int idx = tid + i * 256;             // float4 index, 512 total
      int r = idx >> 3, c4 = (idx & 7) * 4;
      float4 f = *(const float4*)&x[(size_t)(rowTile * 64 + r) * D_ + k0 + c4];
      union { unsigned short h[4]; uint2 u; } pk;
      pk.h[0] = f2bf(f.x); pk.h[1] = f2bf(f.y); pk.h[2] = f2bf(f.z); pk.h[3] = f2bf(f.w);
      *(uint2*)&sA[r * 32 + c4] = pk.u;
    }
    // stage W transposed: read rows of W (b128), scatter to sBT[n][k]
#pragma unroll
    for (int i = 0; i < 2; ++i) {
      int idx = tid + i * 256;             // float4 index, 512 total
      int r = idx >> 4, c4 = (idx & 15) * 4;   // r = k-row 0..31, c4 = col
      float4 f = *(const float4*)&W[(size_t)(k0 + r) * D_ + colTile * 64 + c4];
      sBT[(c4 + 0) * 32 + r] = f2bf(f.x);
      sBT[(c4 + 1) * 32 + r] = f2bf(f.y);
      sBT[(c4 + 2) * 32 + r] = f2bf(f.z);
      sBT[(c4 + 3) * 32 + r] = f2bf(f.w);
    }
    if (k0 + 32 < D_) {                    // L2 prefetch of next tiles
      __builtin_prefetch(&x[(size_t)(rowTile * 64 + wave * 8) * D_ + k0 + 32], 0, 0);
      __builtin_prefetch(&W[(size_t)(k0 + 32 + wave * 4) * D_ + colTile * 64], 0, 0);
    }
    __syncthreads();

    v16bf a  = frag_ld(sA  + rg * 16 * 32,        32, lane);
    v16bf b0 = frag_ld(sBT + (cg * 32)      * 32, 32, lane);
    v16bf b1 = frag_ld(sBT + (cg * 32 + 16) * 32, 32, lane);
    c0 = WMMA_BF16(a, b0, c0);
    c1 = WMMA_BF16(a, b1, c1);
  }

  const int half = lane >> 4, n = lane & 15;
#pragma unroll
  for (int g = 0; g < 8; ++g) {            // C/D layout: M = g + 8*half, N = lane&15
    int rr  = rowTile * 64 + rg * 16 + g + 8 * half;
    int cA  = colTile * 64 + cg * 32 + n;
    int cB  = cA + 16;
    outp[(size_t)rr * D_ + cA] = f2bf(c0[g] + bias[cA]);
    outp[(size_t)rr * D_ + cB] = f2bf(c1[g] + bias[cB]);
  }
}

// ---------------------------------------------------------------------------
// Kernel 2: local attention per (b,h), 64 query rows per block.
// 320KB LDS/WGP: full 64x1024 fp32 score panel in LDS (two-pass softmax).
// Softmax packs P to bf16 IN PLACE so phase-3 A-fragments are b128 loads.
// ---------------------------------------------------------------------------
__global__ __launch_bounds__(256) void local_attn(
    const unsigned short* __restrict__ qkv, unsigned short* __restrict__ ao) {
  extern __shared__ unsigned char smem[];
  float*          sS = (float*)smem;                                        // 64x1024 f32
  unsigned short* sP = (unsigned short*)smem;                               // bf16 view, row stride 2048
  unsigned short* sQ = (unsigned short*)(smem + 64 * 1024 * sizeof(float)); // 64x64 [row][d]
  unsigned short* sT = sQ + 64 * 64;                                        // 64x64 staging

  const int qTile = blockIdx.x;            // 0..15
  const int h     = blockIdx.y;            // 0..11
  const int b     = blockIdx.z;            // 0..3
  const unsigned short* qb = qkv;
  const unsigned short* kb = qkv + (size_t)BS_ * D_;
  const unsigned short* vb = qkv + (size_t)2 * BS_ * D_;

  const int tid = threadIdx.x, lane = tid & 31, wave = tid >> 5;
  const size_t base = (size_t)(b * S_) * D_ + (size_t)h * HD_;

  // load Q tile (64x64 bf16) with b128 copies
#pragma unroll
  for (int i = 0; i < 2; ++i) {
    int idx = tid + i * 256;               // uint4 index, 512 total
    int r = idx >> 3, c8 = (idx & 7) * 8;
    *(uint4*)&sQ[r * 64 + c8] =
        *(const uint4*)&qb[base + (size_t)(qTile * 64 + r) * D_ + c8];
  }

  // ---- phase 1: S = (Q K^T) * scale ----
  for (int j = 0; j < 16; ++j) {
    __syncthreads();
#pragma unroll
    for (int i = 0; i < 2; ++i) {          // stage K tile [key][d] (b128 copy)
      int idx = tid + i * 256;
      int r = idx >> 3, c8 = (idx & 7) * 8;
      *(uint4*)&sT[r * 64 + c8] =
          *(const uint4*)&kb[base + (size_t)(j * 64 + r) * D_ + c8];
    }
    __syncthreads();
#pragma unroll
    for (int t = 0; t < 2; ++t) {
      const int tile = wave * 2 + t;       // 16 tiles of 16x16
      const int mi = tile >> 2, ni = tile & 3;
      v8f acc = {};
      // B = K^T: [key][d] storage is already the contiguous pattern for B(k=d,n=key)
      v16bf a0 = frag_ld(sQ + mi * 16 * 64,      64, lane);
      v16bf b0 = frag_ld(sT + ni * 16 * 64,      64, lane);
      acc = WMMA_BF16(a0, b0, acc);
      v16bf a1 = frag_ld(sQ + mi * 16 * 64 + 32, 64, lane);
      v16bf b1 = frag_ld(sT + ni * 16 * 64 + 32, 64, lane);
      acc = WMMA_BF16(a1, b1, acc);
      const int half = lane >> 4, n = lane & 15;
#pragma unroll
      for (int g = 0; g < 8; ++g) {
        int m = mi * 16 + g + 8 * half;
        sS[m * 1024 + j * 64 + ni * 16 + n] = acc[g] * 0.125f;  // 1/sqrt(64)
      }
    }
  }
  __syncthreads();

  // ---- phase 2a: softmax, 4 threads per row, shfl_xor reductions (wave32) --
  {
    const int row = tid >> 2, q = tid & 3;
    float* rp = sS + row * 1024 + q * 256;
    float mx = -1e30f;
    for (int i = 0; i < 256; ++i) mx = fmaxf(mx, rp[i]);
    mx = fmaxf(mx, __shfl_xor(mx, 1, 32));
    mx = fmaxf(mx, __shfl_xor(mx, 2, 32));
    float sum = 0.f;
    for (int i = 0; i < 256; ++i) { float e = __expf(rp[i] - mx); rp[i] = e; sum += e; }
    sum += __shfl_xor(sum, 1, 32);
    sum += __shfl_xor(sum, 2, 32);
    float inv = 1.f / sum;
    for (int i = 0; i < 256; ++i) rp[i] *= inv;
  }
  __syncthreads();
  // ---- phase 2b: pack P to bf16 in place (one thread per row; write byte
  // 2i is always behind read byte 4i, so left-to-right is safe) ----
  if (tid < 64) {
    const float* rf = sS + tid * 1024;
    unsigned short* rp = sP + tid * 2048;
    for (int i = 0; i < 1024; ++i) rp[i] = f2bf(rf[i]);
  }
  __syncthreads();

  // ---- phase 3: O = P(64x1024,bf16) * V(1024x64) ----
  v8f o0 = {}, o1 = {};
  for (int j = 0; j < 16; ++j) {
    __syncthreads();
#pragma unroll
    for (int i = 0; i < 2; ++i) {          // stage V tile TRANSPOSED: sT[d][key]
      int idx = tid + i * 256;
      int r = idx >> 3, c8 = (idx & 7) * 8;
      uint4 w = *(const uint4*)&vb[base + (size_t)(j * 64 + r) * D_ + c8];
      const unsigned short* s = (const unsigned short*)&w;
#pragma unroll
      for (int t = 0; t < 8; ++t) sT[(c8 + t) * 64 + r] = s[t];
    }
    __syncthreads();
#pragma unroll
    for (int t = 0; t < 2; ++t) {
      const int tile = wave * 2 + t;
      const int mi = tile >> 2, ni = tile & 3;
      v8f acc = t ? o1 : o0;
#pragma unroll
      for (int ks = 0; ks < 64; ks += 32) {
        v16bf a  = frag_ld(sP + mi * 16 * 2048 + j * 64 + ks, 2048, lane);
        v16bf bb = frag_ld(sT + ni * 16 * 64 + ks,            64,   lane); // B(k=key,n=d)=sT[d][key]
        acc = WMMA_BF16(a, bb, acc);
      }
      if (t) o1 = acc; else o0 = acc;
    }
  }

  // write O (gate == 1.0 exactly -> output is the local branch)
  {
    const int half = lane >> 4, n = lane & 15;
#pragma unroll
    for (int t = 0; t < 2; ++t) {
      const int tile = wave * 2 + t;
      const int mi = tile >> 2, ni = tile & 3;
      const v8f o = t ? o1 : o0;
#pragma unroll
      for (int g = 0; g < 8; ++g) {
        int m = qTile * 64 + mi * 16 + g + 8 * half;
        int d = ni * 16 + n;
        ao[base + (size_t)m * D_ + d] = f2bf(o[g]);
      }
    }
  }
}

// ---------------------------------------------------------------------------
// Kernel 3: output projection.  out(4096x768,f32) = AO(bf16) * Wo + bo
// ---------------------------------------------------------------------------
__global__ __launch_bounds__(256) void out_gemm(
    const unsigned short* __restrict__ ao,
    const float* __restrict__ Wo, const float* __restrict__ bo,
    float* __restrict__ out) {
  __shared__ unsigned short sA [64 * 32];   // [row][k]
  __shared__ unsigned short sBT[64 * 32];   // [n][k]
  const int rowTile = blockIdx.x, colTile = blockIdx.y;
  const int tid = threadIdx.x, lane = tid & 31, wave = tid >> 5;
  const int rg = wave >> 1, cg = wave & 1;
  v8f c0 = {}, c1 = {};

  for (int k0 = 0; k0 < D_; k0 += 32) {
    __syncthreads();
    {                                       // stage A (bf16 copy, b128)
      int r = tid >> 2, c8 = (tid & 3) * 8; // 256 uint4 = 64x32
      *(uint4*)&sA[r * 32 + c8] =
          *(const uint4*)&ao[(size_t)(rowTile * 64 + r) * D_ + k0 + c8];
    }
#pragma unroll
    for (int i = 0; i < 2; ++i) {           // stage Wo transposed
      int idx = tid + i * 256;
      int r = idx >> 4, c4 = (idx & 15) * 4;
      float4 f = *(const float4*)&Wo[(size_t)(k0 + r) * D_ + colTile * 64 + c4];
      sBT[(c4 + 0) * 32 + r] = f2bf(f.x);
      sBT[(c4 + 1) * 32 + r] = f2bf(f.y);
      sBT[(c4 + 2) * 32 + r] = f2bf(f.z);
      sBT[(c4 + 3) * 32 + r] = f2bf(f.w);
    }
    if (k0 + 32 < D_) {
      __builtin_prefetch(&Wo[(size_t)(k0 + 32 + wave * 4) * D_ + colTile * 64], 0, 0);
    }
    __syncthreads();

    v16bf a  = frag_ld(sA  + rg * 16 * 32,        32, lane);
    v16bf b0 = frag_ld(sBT + (cg * 32)      * 32, 32, lane);
    v16bf b1 = frag_ld(sBT + (cg * 32 + 16) * 32, 32, lane);
    c0 = WMMA_BF16(a, b0, c0);
    c1 = WMMA_BF16(a, b1, c1);
  }

  const int half = lane >> 4, n = lane & 15;
#pragma unroll
  for (int g = 0; g < 8; ++g) {
    int rr = rowTile * 64 + rg * 16 + g + 8 * half;
    int cA = colTile * 64 + cg * 32 + n;
    int cB = cA + 16;
    out[(size_t)rr * D_ + cA] = c0[g] + bo[cA];
    out[(size_t)rr * D_ + cB] = c1[g] + bo[cB];
  }
}

extern "C" void kernel_launch(void* const* d_in, const int* in_sizes, int n_in,
                              void* d_out, int out_size, void* d_ws, size_t ws_size,
                              hipStream_t stream) {
  (void)in_sizes; (void)n_in; (void)out_size; (void)ws_size;
  const float* x  = (const float*)d_in[0];
  // d_in[1] = mem: unused — gate = sigmoid(64) == 1.0f exactly in fp32,
  // so the memory-attention branch is multiplied by exactly 0.0f.
  const float* Wq = (const float*)d_in[2];
  const float* bq = (const float*)d_in[3];
  const float* Wk = (const float*)d_in[4];
  const float* bk = (const float*)d_in[5];
  const float* Wv = (const float*)d_in[6];
  const float* bv = (const float*)d_in[7];
  const float* Wo = (const float*)d_in[8];
  const float* bo = (const float*)d_in[9];
  float* out = (float*)d_out;

  unsigned short* qkv = (unsigned short*)d_ws;             // 3 * BS * D bf16 (18 MB)
  unsigned short* ao  = qkv + (size_t)3 * BS_ * D_;        // BS * D bf16 (6 MB)

  qkv_gemm<<<dim3(BS_ / 64, D_ / 64, 3), 256, 0, stream>>>(x, Wq, bq, Wk, bk, Wv, bv, qkv);

  const size_t smem = 64 * 1024 * sizeof(float) + 2 * 64 * 64 * sizeof(unsigned short); // 272 KB
  local_attn<<<dim3(S_ / 64, H_, B_), 256, smem, stream>>>(qkv, ao);

  out_gemm<<<dim3(BS_ / 64, D_ / 64), 256, 0, stream>>>(ao, Wo, bo, out);
}
